// Attention_54056458387762
// MI455X (gfx1250) — matrix operand order repaired and checked by reading, not compiled, not run on hardware
//
#include <hip/hip_runtime.h>

// ---------------------------------------------------------------------------
// Fused flash-attention forward: B=32, N=1024, DK=64 (fp32 in/out, u8 mask).
// CDNA5 / gfx1250: wave32, v_wmma_f32_16x16x32_bf16, TDM mask DMA (tensorcnt,
// double-buffered), LDS staging for K/V (bf16), DPP softmax reductions,
// prefetch, exp2-domain softmax.
// ---------------------------------------------------------------------------

typedef __attribute__((ext_vector_type(16))) __bf16        v16bf;
typedef __attribute__((ext_vector_type(8)))  float         v8f;
typedef __attribute__((ext_vector_type(4)))  float         f32x4;
typedef __attribute__((ext_vector_type(4)))  unsigned int  u32x4;
typedef __attribute__((ext_vector_type(2)))  unsigned int  u32x2;
typedef __attribute__((ext_vector_type(8)))  int           i32x8;
typedef __attribute__((ext_vector_type(4)))  int           i32x4;

#define B_    32
#define N_    1024
#define DK_   64
#define KT_   64                  // keys per tile
#define WAVES 8
#define QT_   (WAVES * 16)        // 128 queries per workgroup
#define NTIL  (N_ / KT_)          // 16 key tiles
#define LDK   72                  // padded K/V/P row stride (bf16 elems)
#define MPAD  68                  // padded mask row stride (bytes) = 64 + 1 dword

// dynamic-LDS carve (base offset 0: no static __shared__ in this kernel)
#define KSZ   (KT_ * LDK * 2)             //  9216 B : K tile  [key][d] bf16
#define VSZ   (DK_ * LDK * 2)             //  9216 B : V tile  [d][key] bf16 (transposed)
#define PSZ   (WAVES * 16 * LDK * 2)      // 18432 B : per-wave P buffers
#define MOFF  (KSZ + VSZ + PSZ)           // 36864
#define MSZ   (QT_ * MPAD)                //  8704 B : one mask buffer
#define SMEM_BYTES (MOFF + 2 * MSZ)       // 54272

// TDM availability (ROCm7.2 clang-22: 5-arg; amdgpu-toolchain clang-23: 6-arg)
#ifndef NO_TDM
# if defined(__has_builtin)
#  if __has_builtin(__builtin_amdgcn_tensor_load_to_lds) && \
      __has_builtin(__builtin_amdgcn_s_wait_tensorcnt)
#   define USE_TDM 1
#  endif
# endif
#endif
#ifndef USE_TDM
# define USE_TDM 0
#endif

// DPP butterfly controls (reduction within a 16-lane half)
#define DPP_XOR1  0xB1   // quad_perm [1,0,3,2]
#define DPP_XOR2  0x4E   // quad_perm [2,3,0,1]
#define DPP_HMIR  0x141  // row_half_mirror (^7)
#define DPP_MIR   0x140  // row_mirror      (^15)

template <int CTRL>
__device__ __forceinline__ float dpp_maxf(float x) {
  int y = __builtin_amdgcn_update_dpp(0, __float_as_int(x), CTRL, 0xF, 0xF, true);
  return fmaxf(x, __int_as_float(y));
}
template <int CTRL>
__device__ __forceinline__ float dpp_addf(float x) {
  int y = __builtin_amdgcn_update_dpp(0, __float_as_int(x), CTRL, 0xF, 0xF, true);
  return x + __int_as_float(y);
}
__device__ __forceinline__ float half_reduce_max(float x) {
  x = dpp_maxf<DPP_XOR1>(x);
  x = dpp_maxf<DPP_XOR2>(x);
  x = dpp_maxf<DPP_HMIR>(x);
  x = dpp_maxf<DPP_MIR >(x);
  return x;
}
__device__ __forceinline__ float half_reduce_add(float x) {
  x = dpp_addf<DPP_XOR1>(x);
  x = dpp_addf<DPP_XOR2>(x);
  x = dpp_addf<DPP_HMIR>(x);
  x = dpp_addf<DPP_MIR >(x);
  return x;
}

__device__ __forceinline__ unsigned int pack_bf16x2(float a, float b) {
  union { __bf16 h[2]; unsigned int u; } x;
  x.h[0] = (__bf16)a; x.h[1] = (__bf16)b;
  return x.u;
}

union Frag {                 // one 16x32 bf16 WMMA operand (8 VGPRs / lane)
  v16bf        v;
  unsigned int u[8];
  u32x4        q[2];
};

#if USE_TDM
// One TDM descriptor: 2-D tile (64 cols x 128 rows, 1 B elems) of the mask,
// deposited into LDS with 1 dword of padding after every 16 dwords (row
// stride 68 B) for bank-conflict-free consumption.  Group2/3 unused (2-D).
__device__ __forceinline__ void tdm_mask_tile(const unsigned char* Mg,
                                              int b, int qb, int kt,
                                              unsigned ldsOff) {
  unsigned long long ga = (unsigned long long)(uintptr_t)Mg
      + ((unsigned long long)b * N_ + (unsigned long long)qb * QT_) * N_
      + (unsigned)(kt * KT_);
  u32x4 g0;
  g0.x = 1u;                                    // count=1, user mode, no gather
  g0.y = ldsOff;                                // lds_addr (bytes)
  g0.z = (unsigned)ga;                          // global_addr[31:0]
  g0.w = (unsigned)(ga >> 32) | (2u << 30);     // global_addr[56:32] | type=2
  i32x8 g1;
  g1[0] = (int)((1u << 20) | (3u << 22));       // pad_enable, interval=16dw, amount=1dw
  g1[1] = (int)((unsigned)KT_ << 16);           // tensor_dim0 = 64
  g1[2] = (int)((unsigned)QT_ << 16);           // tensor_dim1 = 128
  g1[3] = (int)((unsigned)KT_ << 16);           // tile_dim0   = 64
  g1[4] = (int)(unsigned)QT_;                   // tile_dim1   = 128, tile_dim2 = 0
  g1[5] = (int)N_;                              // tensor_dim0_stride = 1024
  g1[6] = 0;
  g1[7] = 0;
  i32x4 z4 = {0, 0, 0, 0};
#if __clang_major__ >= 23
  i32x8 z8 = {0, 0, 0, 0, 0, 0, 0, 0};
  __builtin_amdgcn_tensor_load_to_lds(g0, g1, z4, z4, z8, 0);
#else
  __builtin_amdgcn_tensor_load_to_lds(g0, g1, z4, z4, 0);
#endif
}
#endif

__global__ __launch_bounds__(WAVES * 32) void
attn_fa_fwd(const float* __restrict__ Qg, const float* __restrict__ Kg,
            const float* __restrict__ Vg, const unsigned char* __restrict__ Mg,
            float* __restrict__ Og)
{
  extern __shared__ char smem[];
  __bf16* Ksh = (__bf16*)(smem);
  __bf16* Vsh = (__bf16*)(smem + KSZ);
  __bf16* Psh = (__bf16*)(smem + KSZ + VSZ);
  char*   Msh = smem + MOFF;

  const int tid  = threadIdx.x;
  const int lane = tid & 31;
  const int wv   = tid >> 5;
  const int m    = lane & 15;     // tile column / row index
  const int hi   = lane >> 4;     // lane-group half

  const int b  = blockIdx.x >> 3;           // 8 query blocks per batch
  const int qb = blockIdx.x & 7;
  const int q0 = qb * QT_ + wv * 16;        // first query row of this wave

  // ---- Q A-fragments, bf16; fold 1/sqrt(dk) * log2(e) so p = exp2(s - m) --
  Frag aQ[2];
  {
    const float* qp = Qg + ((size_t)b * N_ + (q0 + m)) * DK_;
    const float sc = 0.125f * 1.44269504f;
    #pragma unroll
    for (int f = 0; f < 2; ++f) {
      const int base = f * 32 + hi * 8;     // A layout: K{0..7,16..23}/{8..15,24..31}
      #pragma unroll
      for (int j = 0; j < 4; ++j) {
        aQ[f].u[j]     = pack_bf16x2(qp[base +      2*j] * sc, qp[base +      2*j + 1] * sc);
        aQ[f].u[4 + j] = pack_bf16x2(qp[base + 16 + 2*j] * sc, qp[base + 16 + 2*j + 1] * sc);
      }
    }
  }

  float mI[8], lI[8];
  v8f acc[4];
  #pragma unroll
  for (int r = 0; r < 8; ++r) { mI[r] = -3.0e38f; lI[r] = 0.0f; }
  #pragma unroll
  for (int d = 0; d < 4; ++d) acc[d] = (v8f){0.f,0.f,0.f,0.f,0.f,0.f,0.f,0.f};

  const int srow = tid >> 2;            // K/V staging: row 0..63
  const int c0   = (tid & 3) * 16;      // col base 0/16/32/48

#if USE_TDM
  if (wv == 0) tdm_mask_tile(Mg, b, qb, 0, MOFF);   // prime mask tile 0
#endif

  for (int kt = 0; kt < NTIL; ++kt) {
    __syncthreads();                    // LDS reuse guard

    // ---- stage K tile -> LDS bf16 row-major, V tile -> LDS bf16 transposed
    {
      const size_t rowo = ((size_t)b * N_ + kt * KT_ + srow) * DK_ + c0;
      const float* ks = Kg + rowo;
      const float* vs = Vg + rowo;
      __bf16* kd = Ksh + srow * LDK + c0;
      #pragma unroll
      for (int i = 0; i < 4; ++i) {
        f32x4 f = ((const f32x4*)ks)[i];
        u32x2 w; w.x = pack_bf16x2(f.x, f.y); w.y = pack_bf16x2(f.z, f.w);
        *(u32x2*)(kd + i * 4) = w;
      }
      #pragma unroll
      for (int i = 0; i < 4; ++i) {
        f32x4 f = ((const f32x4*)vs)[i];
        Vsh[(c0 + i*4 + 0) * LDK + srow] = (__bf16)f.x;
        Vsh[(c0 + i*4 + 1) * LDK + srow] = (__bf16)f.y;
        Vsh[(c0 + i*4 + 2) * LDK + srow] = (__bf16)f.z;
        Vsh[(c0 + i*4 + 3) * LDK + srow] = (__bf16)f.w;
      }
      if (kt + 1 < NTIL) {              // global_prefetch_b8 next K/V tile
        __builtin_prefetch(ks + KT_ * DK_, 0, 1);
        __builtin_prefetch(vs + KT_ * DK_, 0, 1);
      }
    }

#if USE_TDM
    // ---- TDM mask pipeline: issue kt+1, guarantee kt complete --------------
    if (wv == 0) {
      if (kt + 1 < NTIL) {
        tdm_mask_tile(Mg, b, qb, kt + 1, MOFF + (unsigned)(((kt + 1) & 1) * MSZ));
        __builtin_amdgcn_s_wait_tensorcnt(1);
      } else {
        __builtin_amdgcn_s_wait_tensorcnt(0);
      }
    }
    const int mcur = kt & 1;
#else
    // ---- fallback: cooperative wide-load mask staging ----------------------
    {
      const int mr = tid >> 1;                   // 0..127
      const int mc = (tid & 1) * 32;
      const u32x4* src = (const u32x4*)(Mg + ((size_t)b * N_ + qb * QT_ + mr) * N_
                                           + kt * KT_ + mc);
      unsigned* dst = (unsigned*)(Msh + mr * MPAD + mc);
      u32x4 w0 = src[0], w1 = src[1];
      dst[0] = w0.x; dst[1] = w0.y; dst[2] = w0.z; dst[3] = w0.w;
      dst[4] = w1.x; dst[5] = w1.y; dst[6] = w1.z; dst[7] = w1.w;
    }
    const int mcur = 0;
#endif
    __syncthreads();

    // ---- S = (Q * scale) K^T : four 16x16 tiles over 64 keys ---------------
    v8f s[4];
    #pragma unroll
    for (int nt = 0; nt < 4; ++nt) {
      Frag bK0, bK1;
      const __bf16* kp = Ksh + (nt * 16 + m) * LDK + hi * 8;  // lane = key col
      bK0.q[0] = *(const u32x4*)(kp);
      bK0.q[1] = *(const u32x4*)(kp + 16);
      bK1.q[0] = *(const u32x4*)(kp + 32);
      bK1.q[1] = *(const u32x4*)(kp + 48);
      v8f c = (v8f){0.f,0.f,0.f,0.f,0.f,0.f,0.f,0.f};
      c = __builtin_amdgcn_wmma_f32_16x16x32_bf16(false, aQ[0].v, false, bK0.v, (short)0, c, false, false);
      c = __builtin_amdgcn_wmma_f32_16x16x32_bf16(false, aQ[1].v, false, bK1.v, (short)0, c, false, false);
      s[nt] = c;
    }

    // ---- mask from LDS: quad-broadcast dword loads + byte test -------------
    {
      const char* mrow = Msh + mcur * MSZ + (wv * 16 + hi * 8) * MPAD + (m & ~3);
      const unsigned msk8 = 0xffu << ((m & 3) * 8);
      #pragma unroll
      for (int nt = 0; nt < 4; ++nt) {
        #pragma unroll
        for (int r = 0; r < 8; ++r) {
          unsigned w = *(const unsigned*)(mrow + r * MPAD + nt * 16);
          if (w & msk8) s[nt][r] = -3.0e38f;
        }
      }
    }

    // ---- online softmax (exp2 domain), DPP row reductions ------------------
    float alpha[8];
    #pragma unroll
    for (int r = 0; r < 8; ++r) {
      float tm = fmaxf(fmaxf(s[0][r], s[1][r]), fmaxf(s[2][r], s[3][r]));
      tm = half_reduce_max(tm);                 // row max within 16-lane half
      const float mN = fmaxf(mI[r], tm);
      alpha[r] = __builtin_amdgcn_exp2f(mI[r] - mN);
      mI[r] = mN;
      float ps = 0.f;
      #pragma unroll
      for (int nt = 0; nt < 4; ++nt) {
        float p = __builtin_amdgcn_exp2f(s[nt][r] - mN);
        s[nt][r] = p;
        ps += p;
      }
      ps = half_reduce_add(ps);                 // row sum within 16-lane half
      lI[r] = lI[r] * alpha[r] + ps;
    }
    #pragma unroll
    for (int d = 0; d < 4; ++d)
      #pragma unroll
      for (int r = 0; r < 8; ++r)
        acc[d][r] *= alpha[r];

    // ---- P: C-layout f32 -> A-layout bf16 via per-wave LDS buffer ----------
    __bf16* pb = Psh + wv * 16 * LDK;
    #pragma unroll
    for (int nt = 0; nt < 4; ++nt)
      #pragma unroll
      for (int r = 0; r < 8; ++r)
        pb[(r + hi * 8) * LDK + nt * 16 + m] = (__bf16)s[nt][r];
    // same-wave DS ops are in-order: no barrier needed before re-reading pb

    Frag aP[2];
    #pragma unroll
    for (int f = 0; f < 2; ++f) {
      const __bf16* pp = pb + m * LDK + f * 32 + hi * 8;
      aP[f].q[0] = *(const u32x4*)pp;
      aP[f].q[1] = *(const u32x4*)(pp + 16);
    }

    // ---- O += P V ----------------------------------------------------------
    #pragma unroll
    for (int dt = 0; dt < 4; ++dt) {
      Frag bV0, bV1;
      const __bf16* vp = Vsh + (dt * 16 + m) * LDK + hi * 8;  // lane = d col
      bV0.q[0] = *(const u32x4*)(vp);
      bV0.q[1] = *(const u32x4*)(vp + 16);
      bV1.q[0] = *(const u32x4*)(vp + 32);
      bV1.q[1] = *(const u32x4*)(vp + 48);
      acc[dt] = __builtin_amdgcn_wmma_f32_16x16x32_bf16(false, aP[0].v, false, bV0.v, (short)0, acc[dt], false, false);
      acc[dt] = __builtin_amdgcn_wmma_f32_16x16x32_bf16(false, aP[1].v, false, bV1.v, (short)0, acc[dt], false, false);
    }
  }

  // ---- normalize and store ------------------------------------------------
  float* op = Og + ((size_t)b * N_ + q0) * DK_;
  #pragma unroll
  for (int r = 0; r < 8; ++r) {
    const float inv = 1.0f / lI[r];
    #pragma unroll
    for (int dt = 0; dt < 4; ++dt)
      op[(size_t)(r + hi * 8) * DK_ + dt * 16 + m] = acc[dt][r] * inv;
  }
}

extern "C" void kernel_launch(void* const* d_in, const int* in_sizes, int n_in,
                              void* d_out, int out_size, void* d_ws, size_t ws_size,
                              hipStream_t stream) {
  (void)in_sizes; (void)n_in; (void)out_size; (void)d_ws; (void)ws_size;
  const float*         Q = (const float*)d_in[0];
  const float*         K = (const float*)d_in[1];
  const float*         V = (const float*)d_in[2];
  const unsigned char* M = (const unsigned char*)d_in[3];  // jnp bool -> 1 byte/elem
  float*               O = (float*)d_out;

  dim3 grid(B_ * (N_ / QT_));    // 32 batches x 8 query-blocks = 256 WGs
  dim3 block(WAVES * 32);        // 8 waves (wave32)
  attn_fa_fwd<<<grid, block, SMEM_BYTES, stream>>>(Q, K, V, M, O);
}
